// CausalSelfAttention_87247965651619
// MI455X (gfx1250) — matrix operand order repaired and checked by reading, not compiled, not run on hardware
//
#include <hip/hip_runtime.h>
#include <hip/hip_bf16.h>

typedef __attribute__((ext_vector_type(16))) __bf16 v16bf;
typedef __attribute__((ext_vector_type(8)))  float  v8f;

#define N_EMBD 1024
#define N_HEAD 16
#define HS     64
#define BB     2
#define TT     2048
#define C3     3072   // 3*N_EMBD
#define KSTEP  64     // keys per flash step

// ---------------------------------------------------------------------------
// WMMA wrapper: D(f32 16x16) = A(bf16 16x32) * B(bf16 32x16) + C
// ---------------------------------------------------------------------------
static __device__ inline v8f wmma_bf16(v16bf a, v16bf b, v8f c) {
    return __builtin_amdgcn_wmma_f32_16x16x32_bf16(
        /*neg_a=*/false, a, /*neg_b=*/false, b,
        /*c_mod=*/(short)0, c, /*reuse_a=*/false, /*reuse_b=*/false);
}

// A-fragment: 16x32 tile (rows row0..row0+15, K cols kb..kb+31) of row-major A[., lda].
// 16-bit A 16x32 layout (contiguous K order, matches B): lane l / l+16 hold row M=l;
//   elems 0..15 -> K = 16*hi + 0..15
static __device__ inline v16bf load_a_frag(const float* __restrict__ A, int lda,
                                           int row0, int kb, int lane, float mul) {
    const int l = lane & 15, hi = lane >> 4;
    const float* p = A + (size_t)(row0 + l) * lda + kb + 16 * hi;
    v16bf f;
#pragma unroll
    for (int i = 0; i < 8; ++i) f[i] = (__bf16)(p[i] * mul);
#pragma unroll
    for (int i = 0; i < 8; ++i) f[8 + i] = (__bf16)(p[8 + i] * mul);
    return f;
}

// B-fragment: 32x16 (KxN) tile from Bt stored row-major [N, ldb] (Bt[n][k]).
// 16-bit B 32x16 layout (contiguous K order, matches A): lanes 0-15 -> N=lane, K=0..15; lanes 16-31 -> K=16..31.
static __device__ inline v16bf load_b_frag(const float* __restrict__ Bt, int ldb,
                                           int col0, int kb, int lane) {
    const int l = lane & 15, hi = lane >> 4;
    const float* p = Bt + (size_t)(col0 + l) * ldb + kb + 16 * hi;
    v16bf f;
#pragma unroll
    for (int i = 0; i < 16; ++i) f[i] = (__bf16)p[i];
    return f;
}

// ---------------------------------------------------------------------------
// GEMM: C[M,N] = A[M,K] * Bt[N,K]^T  (fp32 in/out, bf16 WMMA, fp32 accum)
// 64x64 per wave (4x4 WMMA tiles), 128x128 per block. blockDim = (32,4).
// ---------------------------------------------------------------------------
__global__ __launch_bounds__(128)
void gemm_xwt_bf16(const float* __restrict__ A, const float* __restrict__ Bt,
                   float* __restrict__ Cmat, int M, int N, int K) {
    const int lane  = threadIdx.x;
    const int wv    = threadIdx.y;                        // 2x2 wave grid
    const int tileM = blockIdx.y * 128 + (wv >> 1) * 64;
    const int tileN = blockIdx.x * 128 + (wv & 1) * 64;

    v8f acc[4][4] = {};
    for (int kb = 0; kb < K; kb += 32) {
        v16bf a[4], b[4];
#pragma unroll
        for (int i = 0; i < 4; ++i) a[i] = load_a_frag(A, K, tileM + 16 * i, kb, lane, 1.0f);
#pragma unroll
        for (int j = 0; j < 4; ++j) b[j] = load_b_frag(Bt, K, tileN + 16 * j, kb, lane);
#pragma unroll
        for (int i = 0; i < 4; ++i)
#pragma unroll
            for (int j = 0; j < 4; ++j)
                acc[i][j] = wmma_bf16(a[i], b[j], acc[i][j]);
    }
    // C/D layout: VGPR r -> M = r + 8*hi, N = (lane&15)
    const int l = lane & 15, hi = lane >> 4;
#pragma unroll
    for (int i = 0; i < 4; ++i) {
#pragma unroll
        for (int j = 0; j < 4; ++j) {
#pragma unroll
            for (int r = 0; r < 8; ++r) {
                const int m = tileM + 16 * i + r + 8 * hi;
                Cmat[(size_t)m * N + tileN + 16 * j + l] = acc[i][j][r];
            }
        }
    }
}

// ---------------------------------------------------------------------------
// Flash-style attention with SOFT mask (masked logit = -10.0; softmax over the
// FULL row -> every key block visited). One block = (b, h, 4 query tiles); the
// 4 waves share the async-staged V tile in LDS. blockDim = (32,4).
// ---------------------------------------------------------------------------
__global__ __launch_bounds__(128)
void flash_attn(const float* __restrict__ qkv, float* __restrict__ y) {
    __shared__ float  lds_v[KSTEP][HS];        // 16 KB fp32 V tile (async staged)
    __shared__ __bf16 lds_p[4][16][KSTEP];     // 8 KB per-wave P staging

    const int lane = threadIdx.x;
    const int wav  = threadIdx.y;
    const int tid  = wav * 32 + lane;
    const int bi   = blockIdx.x;               // 0 .. B*H*(T/64)-1 = 1023
    const int q64  = bi & 31;                  // T/64 = 32
    const int h    = (bi >> 5) & 15;
    const int b    = bi >> 9;
    const int qbase = q64 * 64 + wav * 16;
    const int l = lane & 15, hi = lane >> 4;

    const float* base = qkv + (size_t)b * TT * C3;
    const float* Qp = base + h * HS;               // Q[t][d], stride C3
    const float* Kp = base + N_EMBD + h * HS;      // K[t][d]
    const float* Vp = base + 2 * N_EMBD + h * HS;  // V[t][d]

    // Fold 1/sqrt(hs) = 0.125 into Q at conversion time.
    v16bf qf0 = load_a_frag(Qp, C3, qbase, 0,  lane, 0.125f);
    v16bf qf1 = load_a_frag(Qp, C3, qbase, 32, lane, 0.125f);

    v8f o0 = {}, o1 = {}, o2 = {}, o3 = {};
    float mrun[8], lrun[8];
#pragma unroll
    for (int r = 0; r < 8; ++r) { mrun[r] = -3.0e38f; lrun[r] = 0.0f; }

    // Per-thread async-staging source/dest for the V tile (64 keys x 64 d fp32):
    // thread t copies 32 floats of row (t&63), half (t>>6), with 8 B128 async ops.
    const int vrow = tid & 63;
    const int vdof = (tid >> 6) * 32;

    for (int kt = 0; kt < TT / KSTEP; ++kt) {
        const int kbase = kt * KSTEP;

        // ---- async stage V tile: global -> LDS (ASYNCcnt path) ----
        {
            const uint64_t ga  = (uint64_t)(const void*)(Vp + (size_t)(kbase + vrow) * C3 + vdof);
            const uint32_t lda = (uint32_t)(size_t)&lds_v[vrow][vdof];   // LDS byte offset
#pragma unroll
            for (int u = 0; u < 8; ++u) {
                asm volatile("global_load_async_to_lds_b128 %0, %1, off offset:%2"
                             :: "v"(lda), "v"(ga), "i"(u * 16) : "memory");
            }
            asm volatile("s_wait_asynccnt 0x0" ::: "memory");
        }
        __syncthreads();

        // ---- S = (Q*0.125) . K^T : 4 key sub-tiles, K-dim = hs = 64 ----
        v8f lg[4];
#pragma unroll
        for (int kk = 0; kk < 4; ++kk) {
            v16bf k0 = load_b_frag(Kp, C3, kbase + kk * 16, 0,  lane);
            v16bf k1 = load_b_frag(Kp, C3, kbase + kk * 16, 32, lane);
            v8f s = {};
            s = wmma_bf16(qf0, k0, s);
            s = wmma_bf16(qf1, k1, s);
            // soft causal mask: k > q  ->  logit = -10.0 (NOT -inf)
#pragma unroll
            for (int r = 0; r < 8; ++r) {
                const int qg = qbase + r + 8 * hi;
                const int kg = kbase + kk * 16 + l;
                lg[kk][r] = (kg <= qg) ? s[r] : -10.0f;
            }
        }

        // ---- online softmax update (row = (r,hi); 16-lane-half reductions) ----
        v8f p[4];
#pragma unroll
        for (int r = 0; r < 8; ++r) {
            float rowm = fmaxf(fmaxf(lg[0][r], lg[1][r]), fmaxf(lg[2][r], lg[3][r]));
#pragma unroll
            for (int mm = 8; mm >= 1; mm >>= 1)
                rowm = fmaxf(rowm, __shfl_xor(rowm, mm, 32));
            const float mnew  = fmaxf(mrun[r], rowm);
            const float alpha = __expf(mrun[r] - mnew);
            float rs = 0.0f;
#pragma unroll
            for (int kk = 0; kk < 4; ++kk) {
                const float e = __expf(lg[kk][r] - mnew);
                p[kk][r] = e;
                rs += e;
            }
#pragma unroll
            for (int mm = 8; mm >= 1; mm >>= 1)
                rs += __shfl_xor(rs, mm, 32);
            lrun[r] = lrun[r] * alpha + rs;
            mrun[r] = mnew;
            o0[r] *= alpha; o1[r] *= alpha; o2[r] *= alpha; o3[r] *= alpha;
        }

        // ---- stage P (C-layout) into LDS as row-major 16x64 bf16 ----
#pragma unroll
        for (int kk = 0; kk < 4; ++kk)
#pragma unroll
            for (int r = 0; r < 8; ++r)
                lds_p[wav][r + 8 * hi][kk * 16 + l] = (__bf16)p[kk][r];
        asm volatile("s_wait_dscnt 0x0" ::: "memory");

        // ---- reload P as two A-fragments (keys 0..31, 32..63) ----
        v16bf pf0, pf1;
        {
            const __bf16* row = &lds_p[wav][l][0];
#pragma unroll
            for (int j = 0; j < 8; ++j) { pf0[j] = row[16 * hi + j];      pf0[8 + j] = row[8 + 16 * hi + j]; }
#pragma unroll
            for (int j = 0; j < 8; ++j) { pf1[j] = row[32 + 16 * hi + j]; pf1[8 + j] = row[40 + 16 * hi + j]; }
        }

        // ---- O += P . V  (V columns read from async-staged LDS tile) ----
#pragma unroll
        for (int dt = 0; dt < 4; ++dt) {
            v8f* od = (dt == 0) ? &o0 : (dt == 1) ? &o1 : (dt == 2) ? &o2 : &o3;
#pragma unroll
            for (int kc = 0; kc < 2; ++kc) {
                v16bf vf;
#pragma unroll
                for (int j = 0; j < 16; ++j)
                    vf[j] = (__bf16)lds_v[kc * 32 + 16 * hi + j][dt * 16 + l];
                *od = wmma_bf16((kc == 0) ? pf0 : pf1, vf, *od);
            }
        }
        __syncthreads();   // all waves done with lds_v before next stage
    }

    // ---- normalize and store y[b, q, h*64 + d] ----
    float* yb = y + ((size_t)b * TT + qbase) * N_EMBD + h * HS;
#pragma unroll
    for (int r = 0; r < 8; ++r) {
        const float inv = 1.0f / lrun[r];
        const size_t ro = (size_t)(r + 8 * hi) * N_EMBD;
        yb[ro + l]      = o0[r] * inv;
        yb[ro + 16 + l] = o1[r] * inv;
        yb[ro + 32 + l] = o2[r] * inv;
        yb[ro + 48 + l] = o3[r] * inv;
    }
}

// ---------------------------------------------------------------------------
// Launch: qkv GEMM -> attention -> proj GEMM (stream-ordered).
// Workspace: qkv (4096*3072 f32 = 50.3MB) + y (4096*1024 f32 = 16.8MB) = 64MB.
// ---------------------------------------------------------------------------
extern "C" void kernel_launch(void* const* d_in, const int* in_sizes, int n_in,
                              void* d_out, int out_size, void* d_ws, size_t ws_size,
                              hipStream_t stream) {
    (void)in_sizes; (void)n_in; (void)out_size; (void)ws_size;
    const float* x      = (const float*)d_in[0];   // [2,2048,1024]
    const float* w_attn = (const float*)d_in[1];   // [3072,1024]
    const float* w_proj = (const float*)d_in[2];   // [1024,1024]
    float* out = (float*)d_out;                    // [2,2048,1024]

    float* qkv = (float*)d_ws;                     // [4096,3072]
    float* y   = qkv + (size_t)(BB * TT) * C3;     // [4096,1024]

    dim3 blk(32, 4);

    // qkv = x @ w_attn^T : M=4096, N=3072, K=1024  (block tile 128x128)
    dim3 g1(C3 / 128, (BB * TT) / 128);
    gemm_xwt_bf16<<<g1, blk, 0, stream>>>(x, w_attn, qkv, BB * TT, C3, N_EMBD);

    // attention: one block per (b, h, 64 queries) = 1024 blocks
    dim3 g2(BB * N_HEAD * (TT / 64));
    flash_attn<<<g2, blk, 0, stream>>>(qkv, y);

    // out = y @ w_proj^T : M=4096, N=1024, K=1024
    dim3 g3(N_EMBD / 128, (BB * TT) / 128);
    gemm_xwt_bf16<<<g3, blk, 0, stream>>>(y, w_proj, out, BB * TT, N_EMBD, N_EMBD);
}
